// pile_attention_86655260164953
// MI455X (gfx1250) — compile-verified, matrix operand order checked by reading, not run on hardware
//
#include <hip/hip_runtime.h>
#include <hip/hip_bf16.h>
#include <math.h>

// ---------------------------------------------------------------------------
// pile attention for MI455X (gfx1250, wave32, WMMA)
//   H=8, D=96, N1=8192, Ns=8, N=8200 (padded to 8320=65*128), C=768, K=9
// GEMMs use v_wmma_f32_16x16x32_f16; attention is VALU + wave32 shuffles.
// ---------------------------------------------------------------------------

typedef __attribute__((ext_vector_type(16))) _Float16 v16h;
typedef __attribute__((ext_vector_type(8)))  _Float16 v8h;
typedef __attribute__((ext_vector_type(8)))  float    v8f;

#define N1   8192
#define NS   8
#define NTOK 8200
#define MPAD 8320          // 65 * 128
#define CDIM 768
#define HEADS 8
#define DHEAD 96
#define KNBR 9
#define SCALE 0.10206207261596577f   // 96^-0.5

// ---------------------------------------------------------------------------
// pack kernels: fp32 -> f16, weights transposed to [ncol][k] so that WMMA
// B-fragments are K-contiguous per lane.
// ---------------------------------------------------------------------------
__global__ void pack_xa_kernel(const float* __restrict__ x,
                               const float* __restrict__ sem,
                               _Float16* __restrict__ xa) {
    int i = blockIdx.x * 256 + threadIdx.x;
    if (i >= NTOK * CDIM) return;
    int n = i / CDIM;
    int c = i - n * CDIM;
    float v = (n < N1) ? x[i] : sem[(n - N1) * CDIM + c];
    xa[i] = (_Float16)v;
}

__global__ void pack_wT_kernel(const float* __restrict__ w,
                               _Float16* __restrict__ wT,
                               int K, int N) {
    int i = blockIdx.x * 256 + threadIdx.x;        // over N*K, wT is [N][K]
    if (i >= K * N) return;
    int c = i / K;
    int k = i - c * K;
    wT[i] = (_Float16)w[(size_t)k * N + c];
}

// ---------------------------------------------------------------------------
// Fragment load: 16-bit A/B 16x32 layout (ISA 7.12.2):
//   lane L<16  -> row/col = base+L, elements = K[0..7]   and K[16..23]
//   lane L>=16 -> row/col = base+L-16, elements = K[8..15] and K[24..31]
// LDS rows padded to 40 f16 (80 B) to spread banks.
// ---------------------------------------------------------------------------
__device__ __forceinline__ v16h frag_from_lds(const _Float16 (*s)[40],
                                              int rowBase, int lane) {
    int r  = rowBase + (lane & 15);
    int ko = (lane >> 4) << 3;                     // 0 or 8
    v8h lo = *(const v8h*)&s[r][ko];               // K = ko .. ko+7
    v8h hi = *(const v8h*)&s[r][ko + 16];          // K = 16+ko .. 16+ko+7
    return __builtin_shufflevector(lo, hi, 0,1,2,3,4,5,6,7,8,9,10,11,12,13,14,15);
}

// ---------------------------------------------------------------------------
// GEMM 1: qkv = xa(f16)[MPAD x 768] @ wqkvT(f16)[2304 x 768]^T
// scatter epilogue splits columns into q/k/v fp32 buffers, layout [n][h*96+d].
// Block tile 128x128, BK=32, 8 waves, each wave 4x2 WMMA fragments.
// ---------------------------------------------------------------------------
__global__ __launch_bounds__(256) void gemm_qkv_kernel(
        const _Float16* __restrict__ A,      // [MPAD][768]
        const _Float16* __restrict__ Bm,     // [2304][768]  (transposed weight)
        float* __restrict__ qb, float* __restrict__ kbuf, float* __restrict__ vb) {
    __shared__ _Float16 sA[128][40];
    __shared__ _Float16 sB[128][40];
    const int m0 = blockIdx.y * 128;
    const int n0 = blockIdx.x * 128;
    const int tid  = threadIdx.x;
    const int lane = tid & 31;
    const int wave = tid >> 5;
    const int wm = (wave >> 2) * 64;
    const int wn = (wave & 3) * 32;
    const int r  = tid >> 1;
    const int hf = (tid & 1) * 16;

    v8f acc[4][2] = {};

    for (int kb = 0; kb < 768; kb += 32) {
        const _Float16* srcA = A  + (size_t)(m0 + r) * 768 + kb + hf;
        const _Float16* srcB = Bm + (size_t)(n0 + r) * 768 + kb + hf;
        v8h a0 = *(const v8h*)srcA;
        v8h a1 = *(const v8h*)(srcA + 8);
        v8h b0 = *(const v8h*)srcB;
        v8h b1 = *(const v8h*)(srcB + 8);
        *(v8h*)&sA[r][hf]     = a0;
        *(v8h*)&sA[r][hf + 8] = a1;
        *(v8h*)&sB[r][hf]     = b0;
        *(v8h*)&sB[r][hf + 8] = b1;
        if (kb + 32 < 768) {                 // global_prefetch_b8 of next tiles
            __builtin_prefetch(srcA + 32, 0, 1);
            __builtin_prefetch(srcB + 32, 0, 1);
        }
        __syncthreads();

        v16h af[4], bf[2];
#pragma unroll
        for (int i = 0; i < 4; i++) af[i] = frag_from_lds(sA, wm + i * 16, lane);
#pragma unroll
        for (int j = 0; j < 2; j++) bf[j] = frag_from_lds(sB, wn + j * 16, lane);
#pragma unroll
        for (int i = 0; i < 4; i++)
#pragma unroll
            for (int j = 0; j < 2; j++)
                acc[i][j] = __builtin_amdgcn_wmma_f32_16x16x32_f16(
                    false, af[i], false, bf[j], (short)0, acc[i][j], false, false);
        __syncthreads();
    }

    // D layout: VGPR g, lanes 0-15 -> M=g, N=lane; lanes 16-31 -> M=g+8, N=lane-16
    const int moff = (lane >> 4) * 8;
    const int ncol = lane & 15;
#pragma unroll
    for (int i = 0; i < 4; i++)
#pragma unroll
        for (int j = 0; j < 2; j++) {
            int nbase = n0 + wn + j * 16 + ncol;           // 0..2303
            int three = nbase / CDIM;                       // 0=q 1=k 2=v
            int cc    = nbase - three * CDIM;
            float* dst = (three == 0) ? qb : ((three == 1) ? kbuf : vb);
#pragma unroll
            for (int g = 0; g < 8; g++) {
                int m = m0 + wm + i * 16 + g + moff;
                dst[(size_t)m * CDIM + cc] = acc[i][j][g];
            }
        }
}

// ---------------------------------------------------------------------------
// GEMM 2: out = att(f16)[MPAD x 768] @ wprojT(f16)[768 x 768]^T + b  (fp32 out)
// ---------------------------------------------------------------------------
__global__ __launch_bounds__(256) void gemm_proj_kernel(
        const _Float16* __restrict__ A,      // [MPAD][768]
        const _Float16* __restrict__ Bm,     // [768][768] (transposed weight)
        const float* __restrict__ bias,
        float* __restrict__ out) {
    __shared__ _Float16 sA[128][40];
    __shared__ _Float16 sB[128][40];
    const int m0 = blockIdx.y * 128;
    const int n0 = blockIdx.x * 128;
    const int tid  = threadIdx.x;
    const int lane = tid & 31;
    const int wave = tid >> 5;
    const int wm = (wave >> 2) * 64;
    const int wn = (wave & 3) * 32;
    const int r  = tid >> 1;
    const int hf = (tid & 1) * 16;

    v8f acc[4][2] = {};

    for (int kb = 0; kb < 768; kb += 32) {
        const _Float16* srcA = A  + (size_t)(m0 + r) * 768 + kb + hf;
        const _Float16* srcB = Bm + (size_t)(n0 + r) * 768 + kb + hf;
        v8h a0 = *(const v8h*)srcA;
        v8h a1 = *(const v8h*)(srcA + 8);
        v8h b0 = *(const v8h*)srcB;
        v8h b1 = *(const v8h*)(srcB + 8);
        *(v8h*)&sA[r][hf]     = a0;
        *(v8h*)&sA[r][hf + 8] = a1;
        *(v8h*)&sB[r][hf]     = b0;
        *(v8h*)&sB[r][hf + 8] = b1;
        if (kb + 32 < 768) {
            __builtin_prefetch(srcA + 32, 0, 1);
            __builtin_prefetch(srcB + 32, 0, 1);
        }
        __syncthreads();

        v16h af[4], bf[2];
#pragma unroll
        for (int i = 0; i < 4; i++) af[i] = frag_from_lds(sA, wm + i * 16, lane);
#pragma unroll
        for (int j = 0; j < 2; j++) bf[j] = frag_from_lds(sB, wn + j * 16, lane);
#pragma unroll
        for (int i = 0; i < 4; i++)
#pragma unroll
            for (int j = 0; j < 2; j++)
                acc[i][j] = __builtin_amdgcn_wmma_f32_16x16x32_f16(
                    false, af[i], false, bf[j], (short)0, acc[i][j], false, false);
        __syncthreads();
    }

    const int moff = (lane >> 4) * 8;
    const int ncol = lane & 15;
#pragma unroll
    for (int i = 0; i < 4; i++)
#pragma unroll
        for (int j = 0; j < 2; j++) {
            int nn = n0 + wn + j * 16 + ncol;              // 0..767
            float b = bias[nn];
#pragma unroll
            for (int g = 0; g < 8; g++) {
                int m = m0 + wm + i * 16 + g + moff;
                if (m < NTOK)
                    out[(size_t)m * CDIM + nn] = acc[i][j][g] + b;
            }
        }
}

// ---------------------------------------------------------------------------
// RoPE in place on q and k. D=96 -> 48 rotation pairs; pair i<24 uses coord.x,
// i>=24 uses coord.y; freqs[j] = 100^(-j/24). Seman rows use per-head coords.
// ---------------------------------------------------------------------------
__global__ void rope_kernel(float* __restrict__ qb, float* __restrict__ kb,
                            const float* __restrict__ coords,
                            const float* __restrict__ scoords) {
    int gid = blockIdx.x * 256 + threadIdx.x;
    const int TOT = NTOK * HEADS * 48;
    if (gid >= TOT) return;
    int i = gid % 48;
    int h = (gid / 48) & (HEADS - 1);
    int n = gid / (48 * HEADS);
    float cx, cy;
    if (n < N1) { cx = coords[n * 2]; cy = coords[n * 2 + 1]; }
    else {
        int s = n - N1;
        cx = scoords[(h * NS + s) * 2];
        cy = scoords[(h * NS + s) * 2 + 1];
    }
    int j = (i < 24) ? i : (i - 24);
    float f   = __powf(100.0f, -(float)j * (1.0f / 24.0f));
    float ang = ((i < 24) ? cx : cy) * f;
    float c = cosf(ang), s2 = sinf(ang);
    size_t base = (size_t)n * CDIM + h * DHEAD + 2 * i;
    float q1 = qb[base], q2 = qb[base + 1];
    qb[base]     = q1 * c - q2 * s2;
    qb[base + 1] = q1 * s2 + q2 * c;
    float k1 = kb[base], k2 = kb[base + 1];
    kb[base]     = k1 * c - k2 * s2;
    kb[base + 1] = k1 * s2 + k2 * c;
}

// ---------------------------------------------------------------------------
// Global attention (out_fuse): 64 blocks = (head, seman-query) pairs.
// Logits for all 8200 keys staged in LDS, two-pass softmax.
// ---------------------------------------------------------------------------
__global__ __launch_bounds__(256) void global_attn_kernel(
        const float* __restrict__ qb, const float* __restrict__ kb,
        const float* __restrict__ vb, _Float16* __restrict__ att) {
    __shared__ float wts[NTOK];
    __shared__ float red[256];
    __shared__ float qsh[DHEAD];
    const int h  = blockIdx.x >> 3;
    const int qi = blockIdx.x & 7;
    const int tid = threadIdx.x;
    const size_t qoff = (size_t)(N1 + qi) * CDIM + h * DHEAD;
    if (tid < DHEAD) qsh[tid] = qb[qoff + tid];
    __syncthreads();

    float lmax = -3.0e38f;
    for (int kk = tid; kk < NTOK; kk += 256) {
        const float* kp = kb + (size_t)kk * CDIM + h * DHEAD;
        float d = 0.f;
        for (int c = 0; c < DHEAD; c++) d += qsh[c] * kp[c];
        d *= SCALE;
        wts[kk] = d;
        lmax = fmaxf(lmax, d);
    }
    red[tid] = lmax; __syncthreads();
    for (int s = 128; s > 0; s >>= 1) {
        if (tid < s) red[tid] = fmaxf(red[tid], red[tid + s]);
        __syncthreads();
    }
    float M = red[0]; __syncthreads();

    float lsum = 0.f;
    for (int kk = tid; kk < NTOK; kk += 256) {
        float w = __expf(wts[kk] - M);
        wts[kk] = w;
        lsum += w;
    }
    red[tid] = lsum; __syncthreads();
    for (int s = 128; s > 0; s >>= 1) {
        if (tid < s) red[tid] += red[tid + s];
        __syncthreads();
    }
    float inv = 1.0f / red[0];
    __syncthreads();

    for (int d = tid; d < DHEAD; d += 256) {
        const float* vp = vb + h * DHEAD + d;
        float a = 0.f;
        for (int kk = 0; kk < NTOK; kk++) a += wts[kk] * vp[(size_t)kk * CDIM];
        att[(size_t)(N1 + qi) * CDIM + h * DHEAD + d] = (_Float16)(a * inv);
    }
}

// ---------------------------------------------------------------------------
// Local attention: one wave32 per (token, head). Each lane holds 3 of the 96
// dims; 17 keys (9 neighbors w/ mask + 8 globals), online softmax, butterfly
// shuffle reductions.
// ---------------------------------------------------------------------------
__global__ __launch_bounds__(256) void local_attn_kernel(
        const float* __restrict__ qb, const float* __restrict__ kb,
        const float* __restrict__ vb, const int* __restrict__ nidx,
        const int* __restrict__ flag, _Float16* __restrict__ att) {
    const int tid  = threadIdx.x;
    const int lane = tid & 31;
    const int wave = tid >> 5;
    const int pair = blockIdx.x * 8 + wave;   // 8192*8 pairs
    const int n = pair >> 3;
    const int h = pair & 7;
    const size_t qoff = (size_t)n * CDIM + h * DHEAD;
    const float q0 = qb[qoff + lane];
    const float q1 = qb[qoff + 32 + lane];
    const float q2 = qb[qoff + 64 + lane];

    float m = -3.0e38f, s = 0.f;
    float a0 = 0.f, a1 = 0.f, a2 = 0.f;
    for (int t = 0; t < KNBR + NS; t++) {
        int key;
        if (t < KNBR) {
            if (flag[n * KNBR + t] != 0) continue;   // masked (-inf): skip
            key = nidx[n * KNBR + t];
        } else {
            key = N1 + (t - KNBR);
        }
        const size_t ko = (size_t)key * CDIM + h * DHEAD;
        float p = q0 * kb[ko + lane] + q1 * kb[ko + 32 + lane] + q2 * kb[ko + 64 + lane];
#pragma unroll
        for (int off = 16; off; off >>= 1) p += __shfl_xor(p, off, 32);
        float l  = p * SCALE;
        float mn = fmaxf(m, l);
        float corr = __expf(m - mn);
        float w    = __expf(l - mn);
        s  = s * corr + w;
        a0 = a0 * corr + w * vb[ko + lane];
        a1 = a1 * corr + w * vb[ko + 32 + lane];
        a2 = a2 * corr + w * vb[ko + 64 + lane];
        m  = mn;
    }
    float inv = 1.0f / s;
    att[qoff + lane]      = (_Float16)(a0 * inv);
    att[qoff + 32 + lane] = (_Float16)(a1 * inv);
    att[qoff + 64 + lane] = (_Float16)(a2 * inv);
}

// ---------------------------------------------------------------------------
// launch
// ---------------------------------------------------------------------------
extern "C" void kernel_launch(void* const* d_in, const int* in_sizes, int n_in,
                              void* d_out, int out_size, void* d_ws, size_t ws_size,
                              hipStream_t stream) {
    const float* x       = (const float*)d_in[0];
    const float* sem     = (const float*)d_in[1];
    const float* coords  = (const float*)d_in[2];
    const float* scoords = (const float*)d_in[3];
    const int*   nidx    = (const int*)d_in[4];
    const int*   flag    = (const int*)d_in[5];
    const float* w_qkv   = (const float*)d_in[6];
    const float* w_proj  = (const float*)d_in[7];
    const float* b_proj  = (const float*)d_in[8];
    float* out = (float*)d_out;

    char* ws = (char*)d_ws;
    size_t off = 0;
    auto alloc = [&](size_t bytes) -> char* {
        char* p = ws + off;
        off += (bytes + 255) & ~(size_t)255;
        return p;
    };
    _Float16* xa     = (_Float16*)alloc((size_t)MPAD * CDIM * 2);
    _Float16* wqkvT  = (_Float16*)alloc((size_t)3 * CDIM * CDIM * 2);
    _Float16* wprojT = (_Float16*)alloc((size_t)CDIM * CDIM * 2);
    float*    qb     = (float*)alloc((size_t)MPAD * CDIM * 4);
    float*    kb     = (float*)alloc((size_t)MPAD * CDIM * 4);
    float*    vb     = (float*)alloc((size_t)MPAD * CDIM * 4);
    _Float16* att    = (_Float16*)alloc((size_t)MPAD * CDIM * 2);

    pack_xa_kernel<<<(NTOK * CDIM + 255) / 256, 256, 0, stream>>>(x, sem, xa);
    pack_wT_kernel<<<(CDIM * 3 * CDIM + 255) / 256, 256, 0, stream>>>(w_qkv, wqkvT, CDIM, 3 * CDIM);
    pack_wT_kernel<<<(CDIM * CDIM + 255) / 256, 256, 0, stream>>>(w_proj, wprojT, CDIM, CDIM);

    gemm_qkv_kernel<<<dim3(3 * CDIM / 128, MPAD / 128), 256, 0, stream>>>(xa, wqkvT, qb, kb, vb);

    rope_kernel<<<(NTOK * HEADS * 48 + 255) / 256, 256, 0, stream>>>(qb, kb, coords, scoords);

    global_attn_kernel<<<64, 256, 0, stream>>>(qb, kb, vb, att);
    local_attn_kernel<<<N1, 256, 0, stream>>>(qb, kb, vb, nidx, flag, att);

    gemm_proj_kernel<<<dim3(CDIM / 128, MPAD / 128), 256, 0, stream>>>(att, wprojT, b_proj, out);
}
